// Attention_13228499272264
// MI455X (gfx1250) — matrix operand order repaired and checked by reading, not compiled, not run on hardware
//
#include <hip/hip_runtime.h>

// ---------------------------------------------------------------------------
// Causal MHA (B=64,H=4,S=512,Dh=64) + output projection, fp32 in/out.
// Flash-attention with v_wmma_f32_16x16x32_f16 (fp32 accumulate), f16 compute.
// Softmax row reductions use v_permlane16_b32 (VALU) instead of ds_bpermute
// to keep the LDS datapath free for WMMA operand traffic.
// ---------------------------------------------------------------------------

typedef __attribute__((ext_vector_type(16))) _Float16 v16h;
typedef __attribute__((ext_vector_type(8)))  _Float16 v8h;
typedef __attribute__((ext_vector_type(8)))  float    v8f;

constexpr int SEQ = 512;
constexpr int EMB = 256;
constexpr int NH  = 4;
constexpr int DH  = 64;    // head dim
constexpr int QTB = 128;   // q rows per block (stage 1)
constexpr int KCH = 32;    // key chunk (matches WMMA K for f16)

__device__ __forceinline__ v16h cat16(v8h a, v8h b) {
  v16h r;
#pragma unroll
  for (int i = 0; i < 8; ++i) { r[i] = a[i]; r[i + 8] = b[i]; }
  return r;
}

__device__ __forceinline__ v8f wmma16(v16h a, v16h b, v8f c) {
  // D = A(16x32 f16) * B(32x16 f16) + C(16x16 f32)
  return __builtin_amdgcn_wmma_f32_16x16x32_f16(false, a, false, b,
                                                (short)0, c, false, false);
}

// ---- VALU butterfly (xor) within 16-lane rows via v_permlane16_b32 --------
__device__ __forceinline__ float permx(float v, unsigned s0, unsigned s1) {
  int x = __builtin_bit_cast(int, v);
  int r = __builtin_amdgcn_permlane16(x, x, s0, s1, false, false);
  return __builtin_bit_cast(float, r);
}
__device__ __forceinline__ float rowmax16(float v) {
  v = fmaxf(v, permx(v, 0x67452301u, 0xEFCDAB89u)); // xor 1
  v = fmaxf(v, permx(v, 0x54761032u, 0xDCFE98BAu)); // xor 2
  v = fmaxf(v, permx(v, 0x32107654u, 0xBA98FEDCu)); // xor 4
  v = fmaxf(v, permx(v, 0xFEDCBA98u, 0x76543210u)); // xor 8
  return v;
}
__device__ __forceinline__ float rowsum16(float v) {
  v += permx(v, 0x67452301u, 0xEFCDAB89u);
  v += permx(v, 0x54761032u, 0xDCFE98BAu);
  v += permx(v, 0x32107654u, 0xBA98FEDCu);
  v += permx(v, 0xFEDCBA98u, 0x76543210u);
  return v;
}

// ---------------------------------------------------------------------------
// Stage 1: flash attention.  grid = B*H*(SEQ/QTB) blocks, 256 threads (8 waves)
// Each wave: 16 q rows x full head dim (64).  Writes attn output to O (d_out).
// ---------------------------------------------------------------------------
__global__ __launch_bounds__(256, 2)
void attn_fa_kernel(const float* __restrict__ Q, const float* __restrict__ K,
                    const float* __restrict__ V, float* __restrict__ O)
{
  __shared__ _Float16 sK [KCH][DH + 8];        // key chunk, row-major [k][d]
  __shared__ _Float16 sVt[DH][KCH + 8];        // value chunk, transposed [d][k]
  __shared__ _Float16 sP [8][16][KCH + 8];     // per-wave P tile (C->A relayout)

  const int tid  = threadIdx.x;
  const int wave = tid >> 5;
  const int lane = tid & 31;
  const int lh   = lane >> 4;       // lane half (0/1)
  const int ln   = lane & 15;

  const int blk = blockIdx.x;
  const int qb  = blk & 3;          // q tile within sequence
  const int h   = (blk >> 2) & 3;
  const int b   = blk >> 4;

  const int    q0   = qb * QTB + wave * 16;          // first q row of this wave
  const size_t base = ((size_t)b * SEQ) * EMB + (size_t)h * DH;

  const int kbA  = lh * 8;    // unused; A fragments use kb16 too
  const int kb16 = lh * 16;   // B-fragment K base per lane half

  // ---- load Q A-fragments once (scale 1/sqrt(64) folded in) ---------------
  v16h aQ[2];
  {
    const float* qp = Q + base + (size_t)(q0 + ln) * EMB;
#pragma unroll
    for (int f = 0; f < 2; ++f) {
      v16h t;
#pragma unroll
      for (int half = 0; half < 2; ++half) {
        const int d0 = f * 32 + half * 8 + kb16;
        const float4 x0 = *(const float4*)(qp + d0);
        const float4 x1 = *(const float4*)(qp + d0 + 4);
        t[half*8+0] = (_Float16)(x0.x*0.125f); t[half*8+1] = (_Float16)(x0.y*0.125f);
        t[half*8+2] = (_Float16)(x0.z*0.125f); t[half*8+3] = (_Float16)(x0.w*0.125f);
        t[half*8+4] = (_Float16)(x1.x*0.125f); t[half*8+5] = (_Float16)(x1.y*0.125f);
        t[half*8+6] = (_Float16)(x1.z*0.125f); t[half*8+7] = (_Float16)(x1.w*0.125f);
      }
      aQ[f] = t;
    }
  }

  // ---- online-softmax state -----------------------------------------------
  float m_i[8], l_i[8];
  v8f acc[4];
#pragma unroll
  for (int j = 0; j < 8; ++j) { m_i[j] = -3.0e38f; l_i[j] = 0.0f; }
#pragma unroll
  for (int t = 0; t < 4; ++t) acc[t] = (v8f)(0.0f);

  const int kend = qb * QTB + QTB;   // causal: keys < block q upper bound

  for (int kc = 0; kc < kend; kc += KCH) {
    // ---- cooperative K/V chunk load into LDS (fp32 -> f16) ----------------
    {
      const int r  = tid >> 3;            // 0..31 key row
      const int c0 = (tid & 7) * 8;       // 0..56 d col
      const float* kp = K + base + (size_t)(kc + r) * EMB + c0;
      const float* vp = V + base + (size_t)(kc + r) * EMB + c0;
      const float4 k0 = *(const float4*)(kp);
      const float4 k1 = *(const float4*)(kp + 4);
      const float4 v0 = *(const float4*)(vp);
      const float4 v1 = *(const float4*)(vp + 4);
      sK[r][c0+0] = (_Float16)k0.x; sK[r][c0+1] = (_Float16)k0.y;
      sK[r][c0+2] = (_Float16)k0.z; sK[r][c0+3] = (_Float16)k0.w;
      sK[r][c0+4] = (_Float16)k1.x; sK[r][c0+5] = (_Float16)k1.y;
      sK[r][c0+6] = (_Float16)k1.z; sK[r][c0+7] = (_Float16)k1.w;
      sVt[c0+0][r] = (_Float16)v0.x; sVt[c0+1][r] = (_Float16)v0.y;
      sVt[c0+2][r] = (_Float16)v0.z; sVt[c0+3][r] = (_Float16)v0.w;
      sVt[c0+4][r] = (_Float16)v1.x; sVt[c0+5][r] = (_Float16)v1.y;
      sVt[c0+6][r] = (_Float16)v1.z; sVt[c0+7][r] = (_Float16)v1.w;
    }
    __syncthreads();

    if (kc <= q0 + 15) {    // wave-uniform: any unmasked key in this chunk?
      // ---- S = Q K^T : two 16x16 score tiles, 2 WMMAs each ----------------
      v8f s[2];
#pragma unroll
      for (int t2 = 0; t2 < 2; ++t2) {
        const _Float16* kr = &sK[t2 * 16 + ln][0];
        const v16h bk0 = cat16(*(const v8h*)(kr + kb16),
                               *(const v8h*)(kr + kb16 + 8));
        const v16h bk1 = cat16(*(const v8h*)(kr + 32 + kb16),
                               *(const v8h*)(kr + 32 + kb16 + 8));
        v8f c = (v8f)(0.0f);
        c = wmma16(aQ[0], bk0, c);
        c = wmma16(aQ[1], bk1, c);
        s[t2] = c;
      }

      // ---- causal mask + online softmax (rows = (vgpr j, lane half)) ------
      float alpha[8];
#pragma unroll
      for (int j = 0; j < 8; ++j) {
        const int m = j + 8 * lh;
        const int q = q0 + m;
        if (kc + ln      > q) s[0][j] = -1.0e38f;
        if (kc + 16 + ln > q) s[1][j] = -1.0e38f;

        const float rm = rowmax16(fmaxf(s[0][j], s[1][j]));

        const float newm = fmaxf(m_i[j], rm);
        const bool  dead = newm < -1.0e37f;     // entire row still masked
        const float a    = dead ? 1.0f : __expf(m_i[j] - newm);
        const float p0   = dead ? 0.0f : __expf(s[0][j] - newm);
        const float p1   = dead ? 0.0f : __expf(s[1][j] - newm);
        s[0][j] = p0; s[1][j] = p1;

        const float rs = rowsum16(p0 + p1);

        l_i[j] = l_i[j] * a + rs;
        m_i[j] = newm;
        alpha[j] = a;
      }
#pragma unroll
      for (int t2 = 0; t2 < 4; ++t2)
#pragma unroll
        for (int j = 0; j < 8; ++j) acc[t2][j] *= alpha[j];

      // ---- P (C-layout) -> LDS -> A-fragment ------------------------------
#pragma unroll
      for (int j = 0; j < 8; ++j) {
        const int m = j + 8 * lh;
        sP[wave][m][ln]      = (_Float16)s[0][j];
        sP[wave][m][16 + ln] = (_Float16)s[1][j];
      }
      asm volatile("s_wait_dscnt 0" ::: "memory");  // cross-lane LDS RAW (same wave)

      const _Float16* pr = &sP[wave][ln][0];
      const v16h aP = cat16(*(const v8h*)(pr + kb16),
                            *(const v8h*)(pr + kb16 + 8));

      // ---- O += P V : 4 WMMAs over the 64-wide head dim -------------------
#pragma unroll
      for (int dt = 0; dt < 4; ++dt) {
        const _Float16* vr = &sVt[dt * 16 + ln][0];
        const v16h bv = cat16(*(const v8h*)(vr + kb16),
                              *(const v8h*)(vr + kb16 + 8));
        acc[dt] = wmma16(aP, bv, acc[dt]);
      }
    }
    __syncthreads();
  }

  // ---- normalize and store ------------------------------------------------
#pragma unroll
  for (int j = 0; j < 8; ++j) {
    const int m   = j + 8 * lh;
    const float r = 1.0f / l_i[j];
    float* op = O + base + (size_t)(q0 + m) * EMB;
#pragma unroll
    for (int dt = 0; dt < 4; ++dt)
      op[dt * 16 + ln] = acc[dt][j] * r;
  }
}

// ---------------------------------------------------------------------------
// Stage 2: in-place projection  Y = X @ W^T + b  over X=Y=d_out [32768 x 256].
// Row-local: each block stages its 32 rows fully into LDS before overwriting
// them, so in-place is race-free.  8 waves = 2(M) x 4(N); W read from global
// (L2-resident).  32 WMMAs per wave.
// ---------------------------------------------------------------------------
__global__ __launch_bounds__(256, 2)
void proj_kernel(const float* __restrict__ W, const float* __restrict__ bias,
                 float* __restrict__ Y)
{
  __shared__ _Float16 sX[32][EMB + 8];

  const int tid  = threadIdx.x;
  const int wave = tid >> 5;
  const int lane = tid & 31;
  const int lh   = lane >> 4;
  const int ln   = lane & 15;
  const int wM   = wave >> 2;        // 0..1
  const int wN   = wave & 3;         // 0..3
  const int r0   = blockIdx.x * 32;
  const int nb   = wN * 64;

  // stage this block's X rows (read BEFORE any write to Y)
  {
    const int rl = tid >> 3;
    const int e0 = (tid & 7) * 32;
    const float* xp = Y + (size_t)(r0 + rl) * EMB + e0;
#pragma unroll
    for (int i = 0; i < 32; i += 4) {
      const float4 x = *(const float4*)(xp + i);
      sX[rl][e0+i+0] = (_Float16)x.x; sX[rl][e0+i+1] = (_Float16)x.y;
      sX[rl][e0+i+2] = (_Float16)x.z; sX[rl][e0+i+3] = (_Float16)x.w;
    }
  }
  __syncthreads();

  const int kbA  = lh * 8;
  const int kb16 = lh * 16;
  const int rowl = wM * 16 + ln;

  v8f acc[4];
#pragma unroll
  for (int t = 0; t < 4; ++t) acc[t] = (v8f)(0.0f);

#pragma unroll
  for (int kc = 0; kc < EMB; kc += 32) {
    const v16h a = cat16(*(const v8h*)(&sX[rowl][kc + kb16]),
                         *(const v8h*)(&sX[rowl][kc + kb16 + 8]));
#pragma unroll
    for (int dt = 0; dt < 4; ++dt) {
      const float* wr = W + (size_t)(nb + dt * 16 + ln) * EMB + kc + kb16;
      const float4 c0 = *(const float4*)(wr);
      const float4 c1 = *(const float4*)(wr + 4);
      const float4 c2 = *(const float4*)(wr + 8);
      const float4 c3 = *(const float4*)(wr + 12);
      v16h bw;
      bw[0]=(_Float16)c0.x; bw[1]=(_Float16)c0.y; bw[2]=(_Float16)c0.z; bw[3]=(_Float16)c0.w;
      bw[4]=(_Float16)c1.x; bw[5]=(_Float16)c1.y; bw[6]=(_Float16)c1.z; bw[7]=(_Float16)c1.w;
      bw[8]=(_Float16)c2.x; bw[9]=(_Float16)c2.y; bw[10]=(_Float16)c2.z; bw[11]=(_Float16)c2.w;
      bw[12]=(_Float16)c3.x; bw[13]=(_Float16)c3.y; bw[14]=(_Float16)c3.z; bw[15]=(_Float16)c3.w;
      acc[dt] = wmma16(a, bw, acc[dt]);
    }
  }

  // bias + store (rows owned exclusively by this block)
#pragma unroll
  for (int dt = 0; dt < 4; ++dt) {
    const int ng = nb + dt * 16 + ln;
    const float bb = bias[ng];
#pragma unroll
    for (int j = 0; j < 8; ++j) {
      const int m = j + 8 * lh;
      Y[(size_t)(r0 + wM * 16 + m) * EMB + ng] = acc[dt][j] + bb;
    }
  }
}

// ---------------------------------------------------------------------------
extern "C" void kernel_launch(void* const* d_in, const int* in_sizes, int n_in,
                              void* d_out, int out_size, void* d_ws, size_t ws_size,
                              hipStream_t stream) {
  (void)in_sizes; (void)n_in; (void)d_ws; (void)ws_size; (void)out_size;
  const float* q    = (const float*)d_in[0];
  const float* k    = (const float*)d_in[1];
  const float* v    = (const float*)d_in[2];
  const float* w    = (const float*)d_in[3];
  const float* bias = (const float*)d_in[4];
  float* out = (float*)d_out;

  // attention: 64 batches * 4 heads * 4 q-tiles = 1024 blocks
  attn_fa_kernel<<<dim3(64 * NH * (SEQ / QTB)), 256, 0, stream>>>(q, k, v, out);
  // projection (in-place over d_out): 32768 rows / 32 per block
  proj_kernel<<<dim3((64 * SEQ) / 32), 256, 0, stream>>>(w, bias, out);
}
